// PhysicsAttention_83133386981929
// MI455X (gfx1250) — matrix operand order repaired and checked by reading, not compiled, not run on hardware
//
#include <hip/hip_runtime.h>
#include <hip/hip_bf16.h>

// ---------------- problem constants ----------------
#define B_  8
#define T_  8192
#define C_  256
#define H_  8
#define G_  64
#define D_  32
#define TILE_T 64   // token rows per workgroup

typedef __attribute__((ext_vector_type(16))) __bf16 v16bf;
typedef __attribute__((ext_vector_type(8)))  __bf16 v8bf;
typedef __attribute__((ext_vector_type(8)))  float  v8f;

// ---------------- WMMA helpers (wave32, 16x16x32 bf16 -> f32) ----------------
__device__ __forceinline__ v8f wmma_bf16(v16bf a, v16bf b, v8f c) {
  return __builtin_amdgcn_wmma_f32_16x16x32_bf16(
      /*neg_a=*/false, a, /*neg_b=*/false, b,
      /*c_mod=*/(short)0, c, /*reuse_a=*/false, /*reuse_b=*/false);
}

// A fragment (16xK tile, K0..K0+31) from a row-major [M][K] bf16 buffer.
// ISA layout: lane<16 -> row m, K runs {koff..koff+7, koff+16..koff+23}, koff=(lane>>4)*8
__device__ __forceinline__ v16bf load_a_bf(const __bf16* base, int ld, int m0, int k0) {
  int lane = threadIdx.x & 31;
  int m = m0 + (lane & 15);
  int koff = (lane >> 4) * 8;
  const __bf16* p = base + (size_t)m * ld + k0 + koff;
  v8bf lo = *(const v8bf*)p;
  v8bf hi = *(const v8bf*)(p + 16);
  v16bf r;
#pragma unroll
  for (int i = 0; i < 8; ++i) { r[i] = lo[i]; r[i + 8] = hi[i]; }
  return r;
}

// Same A fragment but sourced from fp32 row-major [M][K] global memory (cvt on the fly)
__device__ __forceinline__ v16bf load_a_f32(const float* base, int ld, int m0, int k0) {
  int lane = threadIdx.x & 31;
  int m = m0 + (lane & 15);
  int koff = (lane >> 4) * 8;
  const float* p = base + (size_t)m * ld + k0 + koff;
  v8f lo = *(const v8f*)p;
  v8f hi = *(const v8f*)(p + 16);
  v16bf r;
#pragma unroll
  for (int i = 0; i < 8; ++i) { r[i] = (__bf16)lo[i]; r[i + 8] = (__bf16)hi[i]; }
  return r;
}

// B fragment (Kx16 tile) from an [N][K]-transposed bf16 buffer (row = output col).
// ISA layout: lane -> col n=lane&15, K runs {koff..koff+7, koff+16..koff+23}, koff=(lane>>4)*8
__device__ __forceinline__ v16bf load_b_t(const __bf16* baseNT, int ld, int n0, int k0) {
  int lane = threadIdx.x & 31;
  int n = lane & 15;
  int koff = (lane >> 4) * 8;
  const __bf16* p = baseNT + (size_t)(n0 + n) * ld + k0 + koff;
  v8bf lo = *(const v8bf*)p;
  v8bf hi = *(const v8bf*)(p + 16);
  v16bf r;
#pragma unroll
  for (int i = 0; i < 8; ++i) { r[i] = lo[i]; r[i + 8] = hi[i]; }
  return r;
}

// ---------------- kernel 0: weight transpose->bf16, zero accumulators ----------------
__global__ __launch_bounds__(256) void k0_prep(
    const float* __restrict__ W_in, const float* __restrict__ W_slice,
    const float* __restrict__ W_out,
    __bf16* __restrict__ WtIn, __bf16* __restrict__ WsT, __bf16* __restrict__ WtOut,
    float* __restrict__ sacc, float* __restrict__ snorm) {
  int tid = blockIdx.x * 256 + threadIdx.x;      // grid covers 131072
  if (tid < C_ * C_) {
    int n = tid >> 8, k = tid & 255;             // Wt[n][k] = W[k][n]
    WtIn[tid]  = (__bf16)W_in[k * C_ + n];
    WtOut[tid] = (__bf16)W_out[k * C_ + n];
  }
  if (tid < G_ * D_) {
    int g = tid >> 5, d = tid & 31;              // WsT[g][d] = W_slice[d][g]
    WsT[tid] = (__bf16)W_slice[d * G_ + g];
  }
  if (tid < B_ * H_ * G_ * D_) sacc[tid] = 0.0f;
  if (tid < B_ * H_ * G_) snorm[tid] = 0.0f;
}

// ---------------- kernel 1: in_proj + slice softmax + weighted pooling ----------------
__global__ __launch_bounds__(256) void k1_stageA(
    const float* __restrict__ xq, const float* __restrict__ b_in,
    const float* __restrict__ b_slice, const float* __restrict__ temp,
    const __bf16* __restrict__ WtIn, const __bf16* __restrict__ WsT,
    __bf16* __restrict__ sw, float* __restrict__ sacc, float* __restrict__ snorm) {
  __shared__ __bf16 tmpA[TILE_T * 272];   // tmp[n][c], row-major (A-fragment layout)
  __shared__ __bf16 tmpT[C_ * 80];        // tmp[c][n], transposed (B-fragment layout)
  __shared__ float  lg[TILE_T * 68];      // per-head logits / exp scratch
  __shared__ __bf16 swT[G_ * 80];         // slice weights transposed [g][n]

  const int b  = blockIdx.x >> 7;             // T_/TILE_T = 128 tiles per batch
  const int t0 = (blockIdx.x & 127) * TILE_T;
  const int tid = threadIdx.x, lane = tid & 31, wv = tid >> 5;
  const int rl = lane & 15, rh = lane >> 4;

  // ---- GEMM1: tmp = x_q_tile @ W_in + b_in   (64 x 256, K=256) ----
  const int mstrip = wv & 3, nhalf = wv >> 2;   // wave tile: 16 rows x 128 cols
  const float* xbase = xq + ((size_t)b * T_ + t0) * C_;
  v8f acc[8];
#pragma unroll
  for (int nt = 0; nt < 8; ++nt) acc[nt] = (v8f)(0.0f);
#pragma unroll
  for (int kk = 0; kk < 8; ++kk) {
    const int k0 = kk * 32;
    v16bf a = load_a_f32(xbase, C_, mstrip * 16, k0);
#pragma unroll
    for (int nt = 0; nt < 8; ++nt) {
      v16bf bb = load_b_t(WtIn, C_, nhalf * 128 + nt * 16, k0);
      acc[nt] = wmma_bf16(a, bb, acc[nt]);
    }
  }
#pragma unroll
  for (int nt = 0; nt < 8; ++nt) {
    const int n = nhalf * 128 + nt * 16 + rl;
    const float bias = b_in[n];
#pragma unroll
    for (int j = 0; j < 8; ++j) {
      const int m = mstrip * 16 + rh * 8 + j;
      const __bf16 v = (__bf16)(acc[nt][j] + bias);
      tmpA[m * 272 + n] = v;
      tmpT[n * 80 + m]  = v;
    }
  }
  __syncthreads();

  // ---- per-head: logits -> softmax -> pooled accumulation ----
  for (int h = 0; h < H_; ++h) {
    const int bh = b * H_ + h;
    const float invt = 1.0f / temp[h];

    // GEMM2: logits = tmp_head(64x32) @ W_slice(32x64); 16 tiles, 2 per wave
#pragma unroll
    for (int t = 0; t < 2; ++t) {
      const int ti = wv * 2 + t, mt = ti >> 2, nt = ti & 3;
      v16bf a  = load_a_bf(tmpA, 272, mt * 16, h * 32);
      v16bf bb = load_b_t(WsT, D_, nt * 16, 0);
      v8f c = (v8f)(0.0f);
      c = wmma_bf16(a, bb, c);
      const int g = nt * 16 + rl;
      const float bs = b_slice[g];
#pragma unroll
      for (int j = 0; j < 8; ++j)
        lg[(mt * 16 + rh * 8 + j) * 68 + g] = (c[j] + bs) * invt;
    }
    __syncthreads();

    // row softmax over G, write sw (global bf16) + swT (LDS transposed)
    if (tid < TILE_T) {
      const int n = tid;
      float mx = -1e30f;
      for (int g = 0; g < G_; ++g) mx = fmaxf(mx, lg[n * 68 + g]);
      float s = 0.0f;
      for (int g = 0; g < G_; ++g) { float e = __expf(lg[n * 68 + g] - mx); lg[n * 68 + g] = e; s += e; }
      const float inv = 1.0f / s;
      __bf16* dst = sw + (((size_t)bh * T_) + t0 + n) * G_;
      for (int g = 0; g < G_; ++g) {
        const __bf16 w = (__bf16)(lg[n * 68 + g] * inv);
        swT[g * 80 + n] = w;
        dst[g] = w;
      }
    }
    __syncthreads();

    // slice_norm[g] += sum_n sw[n][g]
    if (tid < G_) {
      float s = 0.0f;
      for (int n = 0; n < TILE_T; ++n) s += (float)swT[tid * 80 + n];
      atomicAdd(&snorm[bh * G_ + tid], s);
    }

    // GEMM3: part(G x D) = sw^T(64x64) @ tmp_head(64x32); 8 tiles, 1 per wave
    {
      const int mt = wv >> 1, nt = wv & 1;
      v8f c = (v8f)(0.0f);
#pragma unroll
      for (int kk = 0; kk < 2; ++kk) {
        const int k0 = kk * 32;
        v16bf a  = load_a_bf(swT, 80, mt * 16, k0);
        v16bf bb = load_b_t(tmpT + (size_t)(h * 32) * 80, 80, nt * 16, k0);
        c = wmma_bf16(a, bb, c);
      }
      float* dst = sacc + ((size_t)bh * G_ + mt * 16 + rh * 8) * D_ + nt * 16 + rl;
#pragma unroll
      for (int j = 0; j < 8; ++j) atomicAdd(dst + (size_t)j * D_, c[j]);
    }
    __syncthreads();   // lg/swT reused next head
  }
}

// ---------------- kernel 2: tokens + QKV + GxG attention (tiny, fp32 VALU) ----------------
__global__ __launch_bounds__(64) void k2_tokens(
    const float* __restrict__ sacc, const float* __restrict__ snorm,
    const float* __restrict__ Wq, const float* __restrict__ bq,
    const float* __restrict__ Wk, const float* __restrict__ bk,
    const float* __restrict__ Wv, const float* __restrict__ bv,
    __bf16* __restrict__ tokT) {
  __shared__ float tok[G_ * 33];
  __shared__ float kl[G_ * 33];
  __shared__ float vl[G_ * 33];
  const int bh = blockIdx.x;
  const int g = threadIdx.x;

  const float nrm = snorm[bh * G_ + g] + 1e-5f;
  const float* ap = sacc + ((size_t)bh * G_ + g) * D_;
#pragma unroll
  for (int d = 0; d < D_; ++d) tok[g * 33 + d] = ap[d] / nrm;
  __syncthreads();

  float q[D_];
#pragma unroll
  for (int d = 0; d < D_; ++d) {
    float aq = bq[d], ak = bk[d], av = bv[d];
#pragma unroll
    for (int e = 0; e < D_; ++e) {
      const float t = tok[g * 33 + e];
      aq += t * Wq[e * D_ + d];
      ak += t * Wk[e * D_ + d];
      av += t * Wv[e * D_ + d];
    }
    q[d] = aq; kl[g * 33 + d] = ak; vl[g * 33 + d] = av;
  }
  __syncthreads();

  const float scale = 0.1767766952966369f;  // 1/sqrt(32)
  float s[G_];
  float mx = -1e30f;
#pragma unroll
  for (int j = 0; j < G_; ++j) {
    float a = 0.0f;
#pragma unroll
    for (int d = 0; d < D_; ++d) a += q[d] * kl[j * 33 + d];
    a *= scale; s[j] = a; mx = fmaxf(mx, a);
  }
  float sum = 0.0f;
#pragma unroll
  for (int j = 0; j < G_; ++j) { const float e = __expf(s[j] - mx); s[j] = e; sum += e; }
  const float inv = 1.0f / sum;
#pragma unroll
  for (int d = 0; d < D_; ++d) {
    float o = 0.0f;
#pragma unroll
    for (int j = 0; j < G_; ++j) o += s[j] * vl[j * 33 + d];
    tokT[((size_t)bh * D_ + d) * G_ + g] = (__bf16)(o * inv);  // transposed [d][g]
  }
}

// ---------------- kernel 3: scatter (sw @ out_tok) + out_proj ----------------
__global__ __launch_bounds__(256) void k3_stageC(
    const __bf16* __restrict__ sw, const __bf16* __restrict__ tokT,
    const __bf16* __restrict__ WtOut, const float* __restrict__ b_out,
    float* __restrict__ out) {
  __shared__ __bf16 yA[TILE_T * 272];     // scattered points [n][c], bf16
  const int b  = blockIdx.x >> 7;
  const int t0 = (blockIdx.x & 127) * TILE_T;
  const int tid = threadIdx.x, lane = tid & 31, wv = tid >> 5;
  const int rl = lane & 15, rh = lane >> 4;

  // per head: up(64 x 32) = sw_tile(64x64) @ out_tok(64x32); 8 tiles, 1 per wave
  for (int h = 0; h < H_; ++h) {
    const int bh = b * H_ + h;
    const __bf16* A  = sw + ((size_t)bh * T_ + t0) * G_;     // [n][g]
    const __bf16* Bt = tokT + (size_t)bh * (D_ * G_);        // [d][g]
    const int mt = wv >> 1, dt = wv & 1;
    v8f c = (v8f)(0.0f);
#pragma unroll
    for (int kk = 0; kk < 2; ++kk) {
      const int k0 = kk * 32;
      v16bf a  = load_a_bf(A, G_, mt * 16, k0);
      v16bf bb = load_b_t(Bt, G_, dt * 16, k0);
      c = wmma_bf16(a, bb, c);
    }
#pragma unroll
    for (int j = 0; j < 8; ++j)
      yA[(mt * 16 + rh * 8 + j) * 272 + h * 32 + dt * 16 + rl] = (__bf16)c[j];
  }
  __syncthreads();

  // final: out = y(64x256) @ W_out(256x256) + b_out
  const int mstrip = wv & 3, nhalf = wv >> 2;
  v8f acc[8];
#pragma unroll
  for (int nt = 0; nt < 8; ++nt) acc[nt] = (v8f)(0.0f);
#pragma unroll
  for (int kk = 0; kk < 8; ++kk) {
    const int k0 = kk * 32;
    v16bf a = load_a_bf(yA, 272, mstrip * 16, k0);
#pragma unroll
    for (int nt = 0; nt < 8; ++nt) {
      v16bf bb = load_b_t(WtOut, C_, nhalf * 128 + nt * 16, k0);
      acc[nt] = wmma_bf16(a, bb, acc[nt]);
    }
  }
#pragma unroll
  for (int nt = 0; nt < 8; ++nt) {
    const int n = nhalf * 128 + nt * 16 + rl;
    const float bias = b_out[n];
    float* dst = out + ((size_t)b * T_ + t0 + mstrip * 16 + rh * 8) * C_ + n;
#pragma unroll
    for (int j = 0; j < 8; ++j) dst[(size_t)j * C_] = acc[nt][j] + bias;
  }
}

// ---------------- launcher ----------------
extern "C" void kernel_launch(void* const* d_in, const int* in_sizes, int n_in,
                              void* d_out, int out_size, void* d_ws, size_t ws_size,
                              hipStream_t stream) {
  const float* x_q     = (const float*)d_in[0];
  const float* W_in    = (const float*)d_in[3];
  const float* b_in    = (const float*)d_in[4];
  const float* W_slice = (const float*)d_in[5];
  const float* b_slice = (const float*)d_in[6];
  const float* temp    = (const float*)d_in[7];
  const float* W_q     = (const float*)d_in[8];
  const float* b_q     = (const float*)d_in[9];
  const float* W_k     = (const float*)d_in[10];
  const float* b_k     = (const float*)d_in[11];
  const float* W_v     = (const float*)d_in[12];
  const float* b_v     = (const float*)d_in[13];
  const float* W_out   = (const float*)d_in[14];
  const float* b_out   = (const float*)d_in[15];
  float* out = (float*)d_out;

  // workspace layout (bytes)
  char* w = (char*)d_ws;
  constexpr size_t SW_BYTES    = (size_t)B_ * H_ * T_ * G_ * 2;   // 64 MB
  constexpr size_t SACC_BYTES  = (size_t)B_ * H_ * G_ * D_ * 4;   // 512 KB
  constexpr size_t SNORM_BYTES = (size_t)B_ * H_ * G_ * 4;        // 16 KB
  constexpr size_t TOKT_BYTES  = (size_t)B_ * H_ * D_ * G_ * 2;   // 256 KB
  constexpr size_t WTIN_BYTES  = (size_t)C_ * C_ * 2;             // 128 KB
  constexpr size_t WST_BYTES   = (size_t)G_ * D_ * 2;             // 4 KB

  __bf16* sw    = (__bf16*)(w);
  float*  sacc  = (float*)(w + SW_BYTES);
  float*  snorm = (float*)(w + SW_BYTES + SACC_BYTES);
  __bf16* tokT  = (__bf16*)(w + SW_BYTES + SACC_BYTES + SNORM_BYTES);
  __bf16* WtIn  = (__bf16*)(w + SW_BYTES + SACC_BYTES + SNORM_BYTES + TOKT_BYTES);
  __bf16* WsT   = (__bf16*)(w + SW_BYTES + SACC_BYTES + SNORM_BYTES + TOKT_BYTES + WTIN_BYTES);
  __bf16* WtOut = (__bf16*)(w + SW_BYTES + SACC_BYTES + SNORM_BYTES + TOKT_BYTES + WTIN_BYTES + WST_BYTES);

  const int tiles = B_ * (T_ / TILE_T);   // 1024

  k0_prep<<<512, 256, 0, stream>>>(W_in, W_slice, W_out, WtIn, WsT, WtOut, sacc, snorm);
  k1_stageA<<<tiles, 256, 0, stream>>>(x_q, b_in, b_slice, temp, WtIn, WsT, sw, sacc, snorm);
  k2_tokens<<<B_ * H_, 64, 0, stream>>>(sacc, snorm, W_q, b_q, W_k, b_k, W_v, b_v, tokT);
  k3_stageC<<<tiles, 256, 0, stream>>>(sw, tokT, WtOut, b_out, out);
}
